// LLaVA_CLIP_5935644803637
// MI455X (gfx1250) — matrix operand order repaired and checked by reading, not compiled, not run on hardware
//
#include <hip/hip_runtime.h>

typedef __attribute__((ext_vector_type(16))) __bf16    v16bf;
typedef __attribute__((ext_vector_type(8)))  float     v8f;
typedef __attribute__((ext_vector_type(8)))  _Float16  h8;   // 16-byte f16 vector

#define BATCH   65536
#define DIM     512
#define HID     1024
#define NCLS    1000
#define NCLSP   1024
#define LN_EPS  1e-5f

// ---------- helpers ----------
__device__ __forceinline__ __bf16 f2bf(float f) {
  unsigned u = __builtin_bit_cast(unsigned, f);
  u += 0x7FFFu + ((u >> 16) & 1u);          // round-to-nearest-even
  unsigned short h = (unsigned short)(u >> 16);
  return __builtin_bit_cast(__bf16, h);
}

__device__ __forceinline__ float quick_gelu(float x) {
  return x / (1.0f + __expf(-1.702f * x));  // x * sigmoid(1.702 x)
}

__device__ __forceinline__ float warp_sum(float v) {
#pragma unroll
  for (int o = 16; o; o >>= 1) v += __shfl_xor(v, o, 32);
  return v;
}

// ---------- f32 -> bf16 conversion (vectorized) ----------
__global__ __launch_bounds__(256) void cvt_bf16_kernel(
    const float* __restrict__ src, __bf16* __restrict__ dst, int n) {
  int i = (blockIdx.x * 256 + threadIdx.x) * 4;
  if (i < n) {
    float4 v = *(const float4*)(src + i);
    dst[i + 0] = f2bf(v.x);
    dst[i + 1] = f2bf(v.y);
    dst[i + 2] = f2bf(v.z);
    dst[i + 3] = f2bf(v.w);
  }
}

// txt [512,1000] f32 -> txtT [1024,512] bf16, rows >= 1000 zero-padded
__global__ __launch_bounds__(256) void txt_transpose_kernel(
    const float* __restrict__ txt, __bf16* __restrict__ dst) {
  int idx = blockIdx.x * 256 + threadIdx.x;   // over 1024*512
  int c = idx >> 9;
  int d = idx & 511;
  float v = (c < NCLS) ? txt[d * NCLS + c] : 0.0f;
  dst[idx] = f2bf(v);
}

// ---------- bf16 WMMA GEMM: out[M,N] = A[M,K] @ W[N,K]^T (+bias, epilogue) ----
// Wave tile 16x128 (8 WMMAs / K-step), 8 waves => block tile 128x128.
// A fragment (16x32 bf16, ISA 7.12.2): lane<16 -> M=lane, K={k0..k0+7, k0+16..23}
//                                      lane>=16 -> M=lane-16, K={k0+8..15, k0+24..31}
// B fragment (32x16 bf16): lane<16 -> N=lane, K={k0..k0+7, k0+16..23} ; lane>=16 -> N=lane-16, K={k0+8..15, k0+24..31}
// All 8 B-fragments live in distinct registers so loads clause up and WMMAs
// use graduated s_wait_loadcnt while later fragments stay in flight.
// Large intermediates are stored f16 (compute stays f32): the pipeline is
// HBM-bound, and f16 storage error (~5e-4) is below the bf16-GEMM noise floor.
union ABits { v16bf v; uint4 q[2]; };

template <int K, int N, int EPI>
__global__ __launch_bounds__(256) void gemm_bf16_wmma_kernel(
    const __bf16* __restrict__ A, const __bf16* __restrict__ W,
    const float* __restrict__ bias, _Float16* __restrict__ outF,
    __bf16* __restrict__ outG) {
  const int wave = threadIdx.x >> 5;
  const int lane = threadIdx.x & 31;
  const int half = lane >> 4;
  const int l16  = lane & 15;
  const int m0 = blockIdx.y * 128 + wave * 16;
  const int n0 = blockIdx.x * 128;

  v8f zero = {0.f, 0.f, 0.f, 0.f, 0.f, 0.f, 0.f, 0.f};
  v8f acc[8];
#pragma unroll
  for (int t = 0; t < 8; ++t) acc[t] = zero;

  const __bf16* aRow  = A + (size_t)(m0 + l16) * K + half * 8;
  const __bf16* wBase = W + (size_t)(n0 + l16) * K + half * 8;

#pragma unroll 2
  for (int k0 = 0; k0 < K; k0 += 32) {
    ABits a;
    ABits b[8];
    // Issue all loads up front (a first, then b0..b7) so wmma[t] only has to
    // wait for its own fragment while later loads remain in flight.
    a.q[0] = *(const uint4*)(aRow + k0);
    a.q[1] = *(const uint4*)(aRow + k0 + 16);
#pragma unroll
    for (int t = 0; t < 8; ++t) {
      const __bf16* bp = wBase + (size_t)(t * 16) * K + k0;
      b[t].q[0] = *(const uint4*)(bp);
      b[t].q[1] = *(const uint4*)(bp + 16);
    }
#pragma unroll
    for (int t = 0; t < 8; ++t) {
      acc[t] = __builtin_amdgcn_wmma_f32_16x16x32_bf16(
          false, a.v, false, b[t].v, (short)0, acc[t], false, false);
    }
  }

  // C/D layout: vgpr r -> row m0 + half*8 + r, col n0 + t*16 + l16
  const int rowBase = m0 + half * 8;
#pragma unroll
  for (int t = 0; t < 8; ++t) {
    const int col = n0 + t * 16 + l16;
    const float bv = (EPI == 2) ? 0.0f : bias[col];
#pragma unroll
    for (int r = 0; r < 8; ++r) {
      size_t o = (size_t)(rowBase + r) * N + col;
      float v = acc[t][r] + bv;
      if (EPI == 0) {            // store pre-act f16 + gelu bf16
        outF[o] = (_Float16)v;
        outG[o] = f2bf(quick_gelu(v));
      } else if (EPI == 1) {     // residual accumulate in-place (f16)
        outF[o] = (_Float16)((float)outF[o] + v);
      } else {                   // plain f16 store
        outF[o] = (_Float16)v;
      }
    }
  }
}

// ---------- LayerNorm over 1024 cols (f16 in) -> bf16 (one wave per row) ----
__global__ __launch_bounds__(256) void ln1024_kernel(
    const _Float16* __restrict__ s, const float* __restrict__ w,
    const float* __restrict__ b, __bf16* __restrict__ out) {
  const int wave = threadIdx.x >> 5;
  const int lane = threadIdx.x & 31;
  const size_t row = (size_t)blockIdx.x * 8 + wave;
  const h8* r8 = (const h8*)(s + row * HID);
  float x[32];
  float sum = 0.f, sq = 0.f;
#pragma unroll
  for (int i = 0; i < 4; ++i) {
    h8 v = r8[lane + i * 32];
#pragma unroll
    for (int e = 0; e < 8; ++e) {
      float f = (float)v[e];
      x[i * 8 + e] = f;
      sum += f;
      sq  += f * f;
    }
  }
  sum = warp_sum(sum);
  sq  = warp_sum(sq);
  const float mu  = sum * (1.0f / HID);
  const float var = sq * (1.0f / HID) - mu * mu;
  const float rs  = rsqrtf(var + LN_EPS);
  __bf16* orow = out + row * HID;
#pragma unroll
  for (int i = 0; i < 4; ++i) {
    int base = (lane + i * 32) * 8;
#pragma unroll
    for (int e = 0; e < 8; ++e) {
      int j = base + e;
      orow[j] = f2bf((x[i * 8 + e] - mu) * rs * w[j] + b[j]);
    }
  }
}

// ---------- LayerNorm(512, f16 in) fused with row-normalize -> bf16 desc ----
__global__ __launch_bounds__(256) void ln512_rownorm_kernel(
    const _Float16* __restrict__ s, const float* __restrict__ w,
    const float* __restrict__ b, __bf16* __restrict__ out) {
  const int wave = threadIdx.x >> 5;
  const int lane = threadIdx.x & 31;
  const size_t row = (size_t)blockIdx.x * 8 + wave;
  const h8* r8 = (const h8*)(s + row * DIM);
  float x[16];
  float sum = 0.f, sq = 0.f;
#pragma unroll
  for (int i = 0; i < 2; ++i) {
    h8 v = r8[lane + i * 32];
#pragma unroll
    for (int e = 0; e < 8; ++e) {
      float f = (float)v[e];
      x[i * 8 + e] = f;
      sum += f;
      sq  += f * f;
    }
  }
  sum = warp_sum(sum);
  sq  = warp_sum(sq);
  const float mu  = sum * (1.0f / DIM);
  const float var = sq * (1.0f / DIM) - mu * mu;
  const float rs  = rsqrtf(var + LN_EPS);
  float y[16];
  float sq2 = 0.f;
#pragma unroll
  for (int i = 0; i < 2; ++i) {
    int base = (lane + i * 32) * 8;
#pragma unroll
    for (int e = 0; e < 8; ++e) {
      int j = base + e;
      float yy = (x[i * 8 + e] - mu) * rs * w[j] + b[j];
      y[i * 8 + e] = yy;
      sq2 += yy * yy;
    }
  }
  const float rn = rsqrtf(warp_sum(sq2));
  __bf16* orow = out + row * DIM;
#pragma unroll
  for (int i = 0; i < 2; ++i) {
    int base = (lane + i * 32) * 8;
#pragma unroll
    for (int e = 0; e < 8; ++e) {
      orow[base + e] = f2bf(y[i * 8 + e] * rn);
    }
  }
}

// ---------- head: norms + combine + argmax + log-softmax loss (per row) -----
__global__ __launch_bounds__(256) void head_kernel(
    const _Float16* __restrict__ sc, const _Float16* __restrict__ sb,
    const float* __restrict__ wp, const float* __restrict__ temp,
    const int* __restrict__ target, float* __restrict__ lossP,
    float* __restrict__ corrP, float* __restrict__ argOut) {
  const int row = blockIdx.x;
  const int tid = threadIdx.x;
  const int w = tid >> 5, l = tid & 31;
  const _Float16* crow = sc + (size_t)row * NCLSP;
  const _Float16* brow = sb + (size_t)row * NCLSP;

  __shared__ float smA[8], smB[8];
  __shared__ float smV[8];
  __shared__ int   smI[8];

  float cv[4], bv[4];
#pragma unroll
  for (int k = 0; k < 4; ++k) {
    int j = tid + k * 256;
    bool ok = j < NCLS;
    cv[k] = ok ? (float)crow[j] : 0.f;
    bv[k] = ok ? (float)brow[j] : 0.f;
  }
  float s1 = 0.f, s2 = 0.f;
#pragma unroll
  for (int k = 0; k < 4; ++k) { s1 += cv[k] * cv[k]; s2 += bv[k] * bv[k]; }
  s1 = warp_sum(s1);
  s2 = warp_sum(s2);
  if (l == 0) { smA[w] = s1; smB[w] = s2; }
  __syncthreads();
  float t1 = 0.f, t2 = 0.f;
#pragma unroll
  for (int i = 0; i < 8; ++i) { t1 += smA[i]; t2 += smB[i]; }
  const float invc = rsqrtf(t1);
  const float invb = rsqrtf(t2);
  const float p = wp[0], q = 1.0f - p;

  float tv[4];
  float s3 = 0.f;
#pragma unroll
  for (int k = 0; k < 4; ++k) {
    tv[k] = p * cv[k] * invc + q * bv[k] * invb;   // zeros stay zero past NCLS
    s3 += tv[k] * tv[k];
  }
  __syncthreads();
  s3 = warp_sum(s3);
  if (l == 0) smA[w] = s3;
  __syncthreads();
  float t3 = 0.f;
#pragma unroll
  for (int i = 0; i < 8; ++i) t3 += smA[i];
  const float invt = rsqrtf(t3);

  // argmax with first-index tie-break
  float m = -3.4e38f;
  int mi = 0;
#pragma unroll
  for (int k = 0; k < 4; ++k) {
    int j = tid + k * 256;
    if (j < NCLS) {
      float lg = tv[k] * invt;
      if (lg > m) { m = lg; mi = j; }
    }
  }
#pragma unroll
  for (int o = 16; o; o >>= 1) {
    float m2 = __shfl_xor(m, o, 32);
    int   i2 = __shfl_xor(mi, o, 32);
    if (m2 > m || (m2 == m && i2 < mi)) { m = m2; mi = i2; }
  }
  __syncthreads();
  if (l == 0) { smV[w] = m; smI[w] = mi; }
  __syncthreads();
  float gm = smV[0];
  int gi = smI[0];
#pragma unroll
  for (int i = 1; i < 8; ++i) {
    if (smV[i] > gm || (smV[i] == gm && smI[i] < gi)) { gm = smV[i]; gi = smI[i]; }
  }

  const float invT = 1.0f / temp[0];
  const int y = target[row];
  float se = 0.f, ly = 0.f;
#pragma unroll
  for (int k = 0; k < 4; ++k) {
    int j = tid + k * 256;
    if (j < NCLS) {
      float lg = tv[k] * invt;
      se += __expf((lg - gm) * invT);
      if (j == y) ly = lg;
    }
  }
  __syncthreads();
  se = warp_sum(se);
  ly = warp_sum(ly);
  if (l == 0) { smA[w] = se; smB[w] = ly; }
  __syncthreads();
  if (tid == 0) {
    float seT = 0.f, lyT = 0.f;
#pragma unroll
    for (int i = 0; i < 8; ++i) { seT += smA[i]; lyT += smB[i]; }
    lossP[row] = (gm - lyT) * invT + __logf(seT);   // -log_softmax[y]
    corrP[row] = (gi == y) ? 1.0f : 0.0f;
    argOut[row] = (float)gi;
  }
}

// ---------- deterministic final reduction (loss mean + accuracy) ----------
__global__ __launch_bounds__(256) void final_reduce_kernel(
    const float* __restrict__ lossP, const float* __restrict__ corrP,
    float* __restrict__ out) {
  float ls = 0.f, cs = 0.f;
  for (int i = threadIdx.x; i < BATCH; i += 256) { ls += lossP[i]; cs += corrP[i]; }
  __shared__ float sl[8], sc2[8];
  ls = warp_sum(ls);
  cs = warp_sum(cs);
  int w = threadIdx.x >> 5;
  if ((threadIdx.x & 31) == 0) { sl[w] = ls; sc2[w] = cs; }
  __syncthreads();
  if (threadIdx.x == 0) {
    float L = 0.f, C2 = 0.f;
    for (int i = 0; i < 8; ++i) { L += sl[i]; C2 += sc2[i]; }
    out[0] = L * (1.0f / (float)BATCH);
    out[1] = C2;
  }
}

// ---------- host launcher ----------
extern "C" void kernel_launch(void* const* d_in, const int* in_sizes, int n_in,
                              void* d_out, int out_size, void* d_ws, size_t ws_size,
                              hipStream_t stream) {
  const float* emb  = (const float*)d_in[0];
  const float* img  = (const float*)d_in[1];
  const float* txt  = (const float*)d_in[2];
  const float* wp   = (const float*)d_in[3];
  const int*   tgt  = (const int*)d_in[4];
  const float* temp = (const float*)d_in[5];
  const float* W0   = (const float*)d_in[6];
  const float* b0   = (const float*)d_in[7];
  const float* W20  = (const float*)d_in[8];
  const float* b20  = (const float*)d_in[9];
  const float* ln0w = (const float*)d_in[10];
  const float* ln0b = (const float*)d_in[11];
  const float* W1   = (const float*)d_in[12];
  const float* b1   = (const float*)d_in[13];
  const float* W21  = (const float*)d_in[14];
  const float* b21  = (const float*)d_in[15];
  const float* ln1w = (const float*)d_in[16];
  const float* ln1b = (const float*)d_in[17];
  // d_in[18], d_in[19] (logit scales) provably cancel under row_norm.

  char* p = (char*)d_ws;
  auto take = [&](size_t bytes) -> char* {
    char* r = p;
    p += (bytes + 255) & ~(size_t)255;
    return r;
  };
  __bf16*    Xbf   = (__bf16*)take((size_t)BATCH * DIM * 2);
  __bf16*    Ibf   = (__bf16*)take((size_t)BATCH * DIM * 2);
  __bf16*    W0bf  = (__bf16*)take((size_t)HID * DIM * 2);
  __bf16*    W20bf = (__bf16*)take((size_t)HID * HID * 2);
  __bf16*    W1bf  = (__bf16*)take((size_t)DIM * HID * 2);
  __bf16*    W21bf = (__bf16*)take((size_t)DIM * DIM * 2);
  __bf16*    Ttbf  = (__bf16*)take((size_t)NCLSP * DIM * 2);
  _Float16*  e1    = (_Float16*)take((size_t)BATCH * HID * 2);
  __bf16*    g1    = (__bf16*)take((size_t)BATCH * HID * 2);
  _Float16*  e1b   = (_Float16*)take((size_t)BATCH * DIM * 2);
  __bf16*    g2    = (__bf16*)take((size_t)BATCH * DIM * 2);
  __bf16*    descb = (__bf16*)take((size_t)BATCH * DIM * 2);
  _Float16*  sbB   = (_Float16*)take((size_t)BATCH * NCLSP * 2);
  _Float16*  scB   = (_Float16*)take((size_t)BATCH * NCLSP * 2);
  float*     lossP = (float*)take((size_t)BATCH * 4);
  float*     corrP = (float*)take((size_t)BATCH * 4);

  dim3 blk(256);
  auto cvt = [&](const float* s, __bf16* d, int n) {
    cvt_bf16_kernel<<<(n / 4 + 255) / 256, blk, 0, stream>>>(s, d, n);
  };
  cvt(emb, Xbf, BATCH * DIM);
  cvt(img, Ibf, BATCH * DIM);
  cvt(W0, W0bf, HID * DIM);
  cvt(W20, W20bf, HID * HID);
  cvt(W1, W1bf, DIM * HID);
  cvt(W21, W21bf, DIM * DIM);
  txt_transpose_kernel<<<(NCLSP * DIM) / 256, blk, 0, stream>>>(txt, Ttbf);

  dim3 gH(HID / 128, BATCH / 128);
  dim3 gD(DIM / 128, BATCH / 128);
  dim3 gC(NCLSP / 128, BATCH / 128);

  // MLP block 0
  gemm_bf16_wmma_kernel<DIM, HID, 0><<<gH, blk, 0, stream>>>(Xbf, W0bf, b0, e1, g1);
  gemm_bf16_wmma_kernel<HID, HID, 1><<<gH, blk, 0, stream>>>(g1, W20bf, b20, e1, nullptr);
  ln1024_kernel<<<BATCH / 8, blk, 0, stream>>>(e1, ln0w, ln0b, g1);
  // MLP block 1
  gemm_bf16_wmma_kernel<HID, DIM, 0><<<gD, blk, 0, stream>>>(g1, W1bf, b1, e1b, g2);
  gemm_bf16_wmma_kernel<DIM, DIM, 1><<<gD, blk, 0, stream>>>(g2, W21bf, b21, e1b, nullptr);
  ln512_rownorm_kernel<<<BATCH / 8, blk, 0, stream>>>(e1b, ln1w, ln1b, descb);
  // similarity GEMMs (padded N=1024, pad cols are exact zeros)
  gemm_bf16_wmma_kernel<DIM, NCLSP, 2><<<gC, blk, 0, stream>>>(descb, Ttbf, nullptr, sbB, nullptr);
  gemm_bf16_wmma_kernel<DIM, NCLSP, 2><<<gC, blk, 0, stream>>>(Ibf, Ttbf, nullptr, scB, nullptr);
  // head + deterministic reduction
  float* outF = (float*)d_out;
  head_kernel<<<BATCH, blk, 0, stream>>>(scB, sbB, wp, temp, tgt, lossP, corrP, outF + 2);
  final_reduce_kernel<<<1, blk, 0, stream>>>(lossP, corrP, outF);
}
